// CausalSelfAttention_5488968204963
// MI455X (gfx1250) — compile-verified, matrix operand order checked by reading, not run on hardware
//
#include <hip/hip_runtime.h>

// ---------------------------------------------------------------------------
// Causal self-attention for MI455X (gfx1250, wave32, WMMA bf16, async->LDS,
// TDM tensor_load_to_lds for attention K/V tiles)
// B=4, T=2048, C=1024, H=16, head_dim=64
// ---------------------------------------------------------------------------

typedef __attribute__((ext_vector_type(16))) __bf16 v16bf;
typedef __attribute__((ext_vector_type(8)))  float  v8f;
typedef __attribute__((ext_vector_type(4)))  unsigned int u32x4;
typedef __attribute__((ext_vector_type(8)))  unsigned int u32x8;

typedef unsigned short u16;
typedef unsigned int   u32;
typedef unsigned long long u64;

#define BB   4
#define TT   2048
#define CC   1024
#define HH   16
#define HD   64
#define MTOT (BB * TT)   // 8192
#define NQKV (3 * CC)    // 3072

__device__ __forceinline__ u16 f2bf(float x) {
    u32 u = __float_as_uint(x);
    u32 r = (u + 0x7FFFu + ((u >> 16) & 1u)) >> 16;   // round-to-nearest-even
    return (u16)r;
}

union FragU { uint4 q[2]; v16bf v; };

// Load a 16x32 bf16 A/B fragment (per-lane: two contiguous 8-element chunks).
// Lane half (0: lanes 0-15, 1: lanes 16-31) selects K base 0 vs 8 per ISA layout.
__device__ __forceinline__ v16bf load_frag(const u16* base, int stride, int row,
                                           int koff, int half) {
    FragU f;
    const u16* p = base + (size_t)row * stride + koff + half * 8;
    f.q[0] = *(const uint4*)(p);
    f.q[1] = *(const uint4*)(p + 16);
    return f.v;
}

__device__ __forceinline__ v8f wmma_bf16(v16bf a, v16bf b, v8f c) {
    return __builtin_amdgcn_wmma_f32_16x16x32_bf16(false, a, false, b,
                                                   (short)0, c, false, false);
}

// ---- CDNA5 async global->LDS copy (ASYNCcnt tracked, bypasses VGPRs) ------
__device__ __forceinline__ void async_ld16(void* lds_ptr, const void* sbase, u32 goff) {
    u32 lds_off = (u32)(size_t)lds_ptr;
    asm volatile("global_load_async_to_lds_b128 %0, %1, %2 offset:0"
                 :: "v"(lds_off), "v"(goff), "s"(sbase)
                 : "memory");
}
template <int N>
__device__ __forceinline__ void wait_async() {
    asm volatile("s_wait_asynccnt %0" :: "i"(N) : "memory");
}
template <int N>
__device__ __forceinline__ void wait_tensor() {
    asm volatile("s_wait_tensorcnt %0" :: "i"(N) : "memory");
}

// ---- CDNA5 Tensor Data Mover: 2-D bf16 tile, global -> LDS ---------------
// D# per CDNA5 ISA. 2-group form (<=2D tensors).
// pad_ic: LDS pad interval code (pads every 2^(pad_ic+1) DWORDs)
// pad_ac: LDS pad amount code (pad_ac+1 DWORDs inserted)
__device__ __forceinline__ void tdm_load_2d(u32 lds_addr, const void* gptr,
                                            u32 tdim0, u32 tdim1, u32 stride0,
                                            u32 tile0, u32 tile1,
                                            u32 pad_ic, u32 pad_ac) {
    u64 ga = (u64)gptr;
    u32x4 g0;
    g0[0] = 1u;                                 // count=1, user mode, no gather
    g0[1] = lds_addr;                           // LDS byte address
    g0[2] = (u32)ga;                            // global_addr[31:0]
    g0[3] = (u32)(ga >> 32) | (2u << 30);       // global_addr[56:32] | type=2
    u32x8 g1;
    g1[0] = (1u << 16)                          // data_size=1 (2 bytes), mask=0
          | (1u << 20)                          // pad_enable
          | (pad_ic << 22) | (pad_ac << 25);
    g1[1] = (tdim0 & 0xFFFFu) << 16;            // barrier_addr=0 | tensor_dim0 lo
    g1[2] = (tdim0 >> 16) | ((tdim1 & 0xFFFFu) << 16);
    g1[3] = (tdim1 >> 16) | (tile0 << 16);      // tensor_dim1 hi | tile_dim0
    g1[4] = tile1;                              // tile_dim1 | tile_dim2=0
    g1[5] = stride0;                            // tensor_dim0_stride[31:0]
    g1[6] = 0;                                  // stride0 hi | stride1 lo
    g1[7] = 0;                                  // stride1 hi
    asm volatile("tensor_load_to_lds %0, %1"
                 :: "s"(g0), "s"(g1)
                 : "memory");
}

// ---------------------------------------------------------------------------
// f32 -> bf16 conversion
// ---------------------------------------------------------------------------
__global__ void __launch_bounds__(256) cvt_f32_bf16(const float* __restrict__ src,
                                                    u16* __restrict__ dst, int n) {
    int i = blockIdx.x * 256 + threadIdx.x;
    if (i < n) dst[i] = f2bf(src[i]);
}

// ---------------------------------------------------------------------------
// NT-GEMM core: out[m,n] = sum_k A[m,k] * W[n,k]   (A,W bf16; f32 accum)
// 128x128 block tile, BK=32, 8 waves, double-buffered async->LDS staging.
// MODE 0: f32 out + bias (proj).  MODE 1: bf16 Q/K/V^T scatter + bias (qkv).
// ---------------------------------------------------------------------------
template <int MODE>
__device__ __forceinline__ void gemm_core(const u16* __restrict__ A,
                                          const u16* __restrict__ W,
                                          const float* __restrict__ bias,
                                          float* __restrict__ outf,
                                          u16* __restrict__ qb,
                                          u16* __restrict__ kb,
                                          u16* __restrict__ vt,
                                          int Ndim) {
    __shared__ __align__(16) u16 As[2][128][48];
    __shared__ __align__(16) u16 Bs[2][128][48];

    const int tid  = threadIdx.x;
    const int wave = tid >> 5, lane = tid & 31;
    const int half = lane >> 4, l16 = lane & 15;
    const int wm = (wave >> 2) * 64;
    const int wn = (wave & 3) * 32;
    const int m0 = blockIdx.y * 128;
    const int n0 = blockIdx.x * 128;

    const int sr0 = tid >> 2;                  // staging row (it adds *64)
    const int sc  = (tid & 3) * 8;             // staging col {0,8,16,24}

    v8f acc[4][2];
#pragma unroll
    for (int i = 0; i < 4; ++i)
#pragma unroll
        for (int j = 0; j < 2; ++j)
            acc[i][j] = (v8f){0.f, 0.f, 0.f, 0.f, 0.f, 0.f, 0.f, 0.f};

    // prologue: stage k-tile 0 into buffer 0
#pragma unroll
    for (int it = 0; it < 2; ++it) {
        int r = sr0 + it * 64;
        async_ld16(&As[0][r][sc], A, (u32)(((m0 + r) * CC + sc) * 2));
        async_ld16(&Bs[0][r][sc], W, (u32)(((n0 + r) * CC + sc) * 2));
    }

    const int KT = CC / 32;
    for (int kt = 0; kt < KT; ++kt) {
        const int buf = kt & 1;
        if (kt + 1 < KT) {
            const int k1 = (kt + 1) * 32;
#pragma unroll
            for (int it = 0; it < 2; ++it) {
                int r = sr0 + it * 64;
                async_ld16(&As[buf ^ 1][r][sc], A, (u32)(((m0 + r) * CC + k1 + sc) * 2));
                async_ld16(&Bs[buf ^ 1][r][sc], W, (u32)(((n0 + r) * CC + k1 + sc) * 2));
            }
            wait_async<4>();    // current tile's 4 copies done (in-order)
        } else {
            wait_async<0>();
        }
        __syncthreads();

        v16bf af[4], bfr[2];
#pragma unroll
        for (int mt = 0; mt < 4; ++mt)
            af[mt] = load_frag(&As[buf][0][0], 48, wm + mt * 16 + l16, 0, half);
#pragma unroll
        for (int nt = 0; nt < 2; ++nt)
            bfr[nt] = load_frag(&Bs[buf][0][0], 48, wn + nt * 16 + l16, 0, half);
#pragma unroll
        for (int mt = 0; mt < 4; ++mt)
#pragma unroll
            for (int nt = 0; nt < 2; ++nt)
                acc[mt][nt] = wmma_bf16(af[mt], bfr[nt], acc[mt][nt]);
        __syncthreads();
    }

    // Epilogue
#pragma unroll
    for (int mt = 0; mt < 4; ++mt) {
#pragma unroll
        for (int nt = 0; nt < 2; ++nt) {
#pragma unroll
            for (int r = 0; r < 8; ++r) {
                int m = m0 + wm + mt * 16 + r + 8 * half;
                int n = n0 + wn + nt * 16 + l16;
                float v = acc[mt][nt][r] + bias[n];
                if (MODE == 0) {
                    outf[(size_t)m * Ndim + n] = v;
                } else {
                    int b   = m >> 11;          // / T
                    int t   = m & (TT - 1);
                    int sec = n >> 10;          // / C : 0=q, 1=k, 2=v
                    int cc  = n & (CC - 1);
                    int h   = cc >> 6;
                    int d   = cc & 63;
                    u16 bv  = f2bf(v);
                    size_t bh = (size_t)(b * HH + h);
                    if (sec == 0)      qb[(bh * TT + t) * HD + d] = bv;
                    else if (sec == 1) kb[(bh * TT + t) * HD + d] = bv;
                    else               vt[(bh * HD + d) * TT + t] = bv;
                }
            }
        }
    }
}

__global__ void __launch_bounds__(256) gemm_qkv(const u16* __restrict__ A,
                                                const u16* __restrict__ W,
                                                const float* __restrict__ bias,
                                                u16* __restrict__ qb,
                                                u16* __restrict__ kb,
                                                u16* __restrict__ vt) {
    gemm_core<1>(A, W, bias, nullptr, qb, kb, vt, NQKV);
}

__global__ void __launch_bounds__(256) gemm_proj(const u16* __restrict__ A,
                                                 const u16* __restrict__ W,
                                                 const float* __restrict__ bias,
                                                 float* __restrict__ out) {
    gemm_core<0>(A, W, bias, out, nullptr, nullptr, nullptr, CC);
}

// ---------------------------------------------------------------------------
// Flash attention: one workgroup (4 waves) per (b*h, 64-row Q block).
// K / V^T blocks staged by the Tensor Data Mover (one descriptor per tile,
// issued by wave 0, TENSORcnt-tracked, TDM applies the padded LDS stride),
// double-buffered across the j loop.
// ---------------------------------------------------------------------------
__global__ void __launch_bounds__(128) attn_fa(const u16* __restrict__ qb,
                                               const u16* __restrict__ kb,
                                               const u16* __restrict__ vt,
                                               u16* __restrict__ yb) {
    __shared__ __align__(16) u16 Ks[2][64][72];   // K block  [tok][hd]   (TDM-padded)
    __shared__ __align__(16) u16 Vs[2][64][72];   // V^T block [hd][tok]  (TDM-padded)
    __shared__ __align__(16) u16 Pl[4][16][64];   // per-wave P staging

    const int tid  = threadIdx.x;
    const int wave = tid >> 5, lane = tid & 31;
    const int half = lane >> 4, l16 = lane & 15;
    const int bh = blockIdx.y;
    const int q0 = blockIdx.x * 64;

    const u16* qp = qb + (size_t)bh * TT * HD;
    const u16* kp = kb + (size_t)bh * TT * HD;
    const u16* vp = vt + (size_t)bh * HD * TT;

    // Q fragments stay in registers
    v16bf aq[2];
#pragma unroll
    for (int kk = 0; kk < 2; ++kk)
        aq[kk] = load_frag(qp, HD, q0 + wave * 16 + l16, kk * 32, half);

    v8f oacc[4];
#pragma unroll
    for (int nt = 0; nt < 4; ++nt)
        oacc[nt] = (v8f){0.f, 0.f, 0.f, 0.f, 0.f, 0.f, 0.f, 0.f};

    float mstat[8], lstat[8];
#pragma unroll
    for (int r = 0; r < 8; ++r) { mstat[r] = -3.0e38f; lstat[r] = 0.f; }

    const int jend = blockIdx.x;

    // TDM tile parameters: 64x64 bf16 tile, 128B rows -> pad interval code 4
    // (32 DWORDs), pad amount code 3 (4 DWORDs) => LDS row stride 72 elements.
    if (wave == 0) {
        tdm_load_2d((u32)(size_t)&Ks[0][0][0], kp, HD, TT, HD, 64, 64, 4, 3);
        tdm_load_2d((u32)(size_t)&Vs[0][0][0], vp, TT, HD, TT, 64, 64, 4, 3);
    }

    for (int j = 0; j <= jend; ++j) {
        const int buf = j & 1;
        if (wave == 0) {
            if (j < jend) {
                const int jn = (j + 1) * 64;
                tdm_load_2d((u32)(size_t)&Ks[buf ^ 1][0][0], kp + (size_t)jn * HD,
                            HD, TT, HD, 64, 64, 4, 3);
                tdm_load_2d((u32)(size_t)&Vs[buf ^ 1][0][0], vp + (size_t)jn,
                            TT, HD, TT, 64, 64, 4, 3);
                wait_tensor<2>();   // current buffer's two descriptors done (in-order)
            } else {
                wait_tensor<0>();
            }
        }
        __syncthreads();

        // ---- S = Q K^T (16x64 per wave) ----
        v8f sacc[4];
#pragma unroll
        for (int nt = 0; nt < 4; ++nt)
            sacc[nt] = (v8f){0.f, 0.f, 0.f, 0.f, 0.f, 0.f, 0.f, 0.f};
#pragma unroll
        for (int kk = 0; kk < 2; ++kk) {
#pragma unroll
            for (int nt = 0; nt < 4; ++nt) {
                v16bf bk = load_frag(&Ks[buf][0][0], 72, nt * 16 + l16, kk * 32, half);
                sacc[nt] = wmma_bf16(aq[kk], bk, sacc[nt]);
            }
        }

        // ---- scale + causal mask ----
        float p[4][8];
        const bool diag = (j == jend);
#pragma unroll
        for (int nt = 0; nt < 4; ++nt) {
#pragma unroll
            for (int r = 0; r < 8; ++r) {
                float sv = sacc[nt][r] * 0.125f;   // 1/sqrt(64)
                if (diag) {
                    int col = j * 64 + nt * 16 + l16;
                    int row = q0 + wave * 16 + r + 8 * half;
                    if (col > row) sv = -3.0e38f;
                }
                p[nt][r] = sv;
            }
        }

        // ---- online softmax (row stats redundant in each 16-lane half) ----
        float mnew[8];
#pragma unroll
        for (int r = 0; r < 8; ++r) {
            float v = fmaxf(fmaxf(p[0][r], p[1][r]), fmaxf(p[2][r], p[3][r]));
            v = fmaxf(v, __shfl_xor(v, 1, 16));
            v = fmaxf(v, __shfl_xor(v, 2, 16));
            v = fmaxf(v, __shfl_xor(v, 4, 16));
            v = fmaxf(v, __shfl_xor(v, 8, 16));
            mnew[r] = fmaxf(mstat[r], v);
        }
#pragma unroll
        for (int nt = 0; nt < 4; ++nt)
#pragma unroll
            for (int r = 0; r < 8; ++r)
                p[nt][r] = __expf(p[nt][r] - mnew[r]);
#pragma unroll
        for (int r = 0; r < 8; ++r) {
            float s = p[0][r] + p[1][r] + p[2][r] + p[3][r];
            s += __shfl_xor(s, 1, 16);
            s += __shfl_xor(s, 2, 16);
            s += __shfl_xor(s, 4, 16);
            s += __shfl_xor(s, 8, 16);
            float al = __expf(mstat[r] - mnew[r]);
            lstat[r] = lstat[r] * al + s;
            mstat[r] = mnew[r];
#pragma unroll
            for (int nt = 0; nt < 4; ++nt)
                oacc[nt][r] = oacc[nt][r] * al;
        }

        // ---- P: C-layout regs -> LDS -> A-layout fragments ----
#pragma unroll
        for (int nt = 0; nt < 4; ++nt)
#pragma unroll
            for (int r = 0; r < 8; ++r)
                Pl[wave][r + 8 * half][nt * 16 + l16] = f2bf(p[nt][r]);

        // ---- O += P V  (V^T from LDS, K-contiguous) ----
#pragma unroll
        for (int kk = 0; kk < 2; ++kk) {
            v16bf ap = load_frag(&Pl[wave][0][0], 64, l16, kk * 32, half);
#pragma unroll
            for (int nt = 0; nt < 4; ++nt) {
                v16bf bv = load_frag(&Vs[buf][0][0], 72, nt * 16 + l16, kk * 32, half);
                oacc[nt] = wmma_bf16(ap, bv, oacc[nt]);
            }
        }
        __syncthreads();   // protect buf from next iteration's TDM writes
    }

    // ---- write y (bf16, [B,T,C] layout) ----
    const int b = bh / HH, h = bh % HH;
#pragma unroll
    for (int nt = 0; nt < 4; ++nt) {
#pragma unroll
        for (int r = 0; r < 8; ++r) {
            int trow = q0 + wave * 16 + r + 8 * half;
            float v = oacc[nt][r] / lstat[r];
            yb[((size_t)b * TT + trow) * CC + h * HD + nt * 16 + l16] = f2bf(v);
        }
    }
}

// ---------------------------------------------------------------------------
// Launcher
// ---------------------------------------------------------------------------
extern "C" void kernel_launch(void* const* d_in, const int* in_sizes, int n_in,
                              void* d_out, int out_size, void* d_ws, size_t ws_size,
                              hipStream_t stream) {
    const float* x  = (const float*)d_in[0];   // [B,T,C]
    const float* wa = (const float*)d_in[1];   // [3C,C]
    const float* ba = (const float*)d_in[2];   // [3C]
    const float* wp = (const float*)d_in[3];   // [C,C]
    const float* bp = (const float*)d_in[4];   // [C]
    float* out = (float*)d_out;                // [B,T,C] f32

    char* ws = (char*)d_ws;
    size_t o = 0;
    u16* xb  = (u16*)(ws + o); o += (size_t)MTOT * CC * 2;   // x bf16
    u16* wab = (u16*)(ws + o); o += (size_t)NQKV * CC * 2;   // w_attn bf16
    u16* wpb = (u16*)(ws + o); o += (size_t)CC * CC * 2;     // w_proj bf16
    u16* qbf = (u16*)(ws + o); o += (size_t)MTOT * CC * 2;   // Q [B,H,T,64]
    u16* kbf = (u16*)(ws + o); o += (size_t)MTOT * CC * 2;   // K [B,H,T,64]
    u16* vtb = (u16*)(ws + o); o += (size_t)MTOT * CC * 2;   // V^T [B,H,64,T]
    u16* yb  = (u16*)(ws + o); o += (size_t)MTOT * CC * 2;   // y [B,T,C]

    int nx = MTOT * CC;
    cvt_f32_bf16<<<(nx + 255) / 256, 256, 0, stream>>>(x, xb, nx);
    int nwa = NQKV * CC;
    cvt_f32_bf16<<<(nwa + 255) / 256, 256, 0, stream>>>(wa, wab, nwa);
    int nwp = CC * CC;
    cvt_f32_bf16<<<(nwp + 255) / 256, 256, 0, stream>>>(wp, wpb, nwp);

    gemm_qkv<<<dim3(NQKV / 128, MTOT / 128), 256, 0, stream>>>(xb, wab, ba, qbf, kbf, vtb);
    attn_fa<<<dim3(TT / 64, BB * HH), 128, 0, stream>>>(qbf, kbf, vtb, yb);
    gemm_proj<<<dim3(CC / 128, MTOT / 128), 256, 0, stream>>>(yb, wpb, bp, out);
}
